// CrossAttention_56212531970722
// MI455X (gfx1250) — matrix operand order repaired and checked by reading, not compiled, not run on hardware
//
#include <hip/hip_runtime.h>
#include <hip/hip_bf16.h>
#include <math.h>

// Problem constants (match reference)
#define B_     8
#define N_     16384
#define D_     128
#define TWO_D  256
#define LN_EPS 1e-5f
#define NCHUNK 64
#define CHUNK  (N_ / NCHUNK)   // 256 rows per stats chunk

typedef __attribute__((ext_vector_type(16))) _Float16 v16h;
typedef __attribute__((ext_vector_type(8)))  float    v8f;

// Async global->LDS copy of one 16-byte chunk (GVS mode: sgpr base + vgpr offset).
// Tracked by ASYNCcnt; consumer must s_wait_asynccnt then barrier.
__device__ __forceinline__ void async_copy_b128(uint32_t lds_addr, uint32_t goff,
                                                const void* base) {
  asm volatile("global_load_async_to_lds_b128 %0, %1, %2"
               :: "v"(lds_addr), "v"(goff), "s"(base)
               : "memory");
}
__device__ __forceinline__ void wait_asynccnt0() {
  asm volatile("s_wait_asynccnt 0" ::: "memory");
}

// ---------------------------------------------------------------------------
// K1a: partial online-softmax stats over N for keys: per (b, chunk, d)
// ---------------------------------------------------------------------------
__global__ __launch_bounds__(128)
void k_colstats_part(const float* __restrict__ x1,
                     float* __restrict__ pmax, float* __restrict__ psum) {
  const int d = threadIdx.x;
  const int chunk = blockIdx.x;
  const int b = blockIdx.y;
  float m = -INFINITY, s = 0.f;
  const float* p = x1 + ((size_t)b * N_ + (size_t)chunk * CHUNK) * D_ + d;
  for (int r = 0; r < CHUNK; ++r) {
    float v = p[(size_t)r * D_];
    if (v > m) { s *= __expf(m - v); m = v; }
    s += __expf(v - m);
  }
  const int o = (b * NCHUNK + chunk) * D_ + d;
  pmax[o] = m;
  psum[o] = s;
}

// ---------------------------------------------------------------------------
// K1b: combine partials -> colmax[b,d], colsinv[b,d] = 1/sum(exp)
// ---------------------------------------------------------------------------
__global__ __launch_bounds__(128)
void k_colstats_final(const float* __restrict__ pmax, const float* __restrict__ psum,
                      float* __restrict__ colmax, float* __restrict__ colsinv) {
  const int d = threadIdx.x;
  const int b = blockIdx.x;
  float m = -INFINITY;
  for (int c = 0; c < NCHUNK; ++c)
    m = fmaxf(m, pmax[(b * NCHUNK + c) * D_ + d]);
  float s = 0.f;
  for (int c = 0; c < NCHUNK; ++c)
    s += psum[(b * NCHUNK + c) * D_ + d] * __expf(pmax[(b * NCHUNK + c) * D_ + d] - m);
  colmax[b * D_ + d]  = m;
  colsinv[b * D_ + d] = 1.0f / s;
}

// ---------------------------------------------------------------------------
// K2: context[b] (128x128) = softmax_N(x1^T) @ x2  via v_wmma_f32_16x16x32_f16
// x2 tile staged by GLOBAL_LOAD_ASYNC_TO_LDS_B128 (ASYNCcnt path); x1 tile
// goes through VALU for the fused exp(x - colmax) and is stored transposed.
// ---------------------------------------------------------------------------
__global__ __launch_bounds__(256)
void k_context_wmma(const float* __restrict__ x1, const float* __restrict__ x2,
                    const float* __restrict__ colmax, const float* __restrict__ colsinv,
                    float* __restrict__ ctxT) {
  __shared__ float sA[D_][33];                   // exp(x1) transposed: [d][k]
  __shared__ __align__(16) float sB[32][D_];     // x2 tile (async dest): [k][v]
  __shared__ float smax[D_], sinv[D_];

  const int b    = blockIdx.x;
  const int tid  = threadIdx.x;
  const int wave = tid >> 5;
  const int lane = tid & 31;
  const int h    = lane >> 4;
  const int mrow = 16 * wave + (lane & 15);

  const uint32_t sB_base = (uint32_t)(uintptr_t)&sB[0][0];

  if (tid < D_) { smax[tid] = colmax[b * D_ + tid]; sinv[tid] = colsinv[b * D_ + tid]; }
  __syncthreads();

  v8f acc[8] = {};

  for (int k0 = 0; k0 < N_; k0 += 32) {
    // async-stage x2 tile: 32x128 f32 = 16KB contiguous = 1024 x b128 chunks
    const uint32_t gbase = (uint32_t)(((size_t)b * N_ + k0) * D_ * sizeof(float));
#pragma unroll
    for (int j = 0; j < 4; ++j) {
      const uint32_t c16 = (uint32_t)tid + 256u * j;
      async_copy_b128(sB_base + 16u * c16, gbase + 16u * c16, x2);
    }
    // regular path: exp(x1 - colmax), transposed scatter into LDS
    for (int idx = tid; idx < 32 * D_; idx += 256) {
      const int r = idx >> 7, c = idx & (D_ - 1);
      sA[c][r] = __expf(x1[((size_t)b * N_ + k0 + r) * D_ + c] - smax[c]);
    }
    wait_asynccnt0();
    __syncthreads();

    // A operand: 16x32 f16: lane L: m=L&15, k=i+16h
    v16h af;
#pragma unroll
    for (int i = 0; i < 16; ++i) {
      const int k = i + 16 * h;
      af[i] = (_Float16)sA[mrow][k];
    }
#pragma unroll
    for (int t = 0; t < 8; ++t) {
      // B operand: 32x16: lane gives col, VGPR j -> k=2j,2j+1 (+16 for h=1)
      v16h bf;
#pragma unroll
      for (int i = 0; i < 16; ++i)
        bf[i] = (_Float16)sB[i + 16 * h][16 * t + (lane & 15)];
      acc[t] = __builtin_amdgcn_wmma_f32_16x16x32_f16(
          false, af, false, bf, (short)0, acc[t], false, false);
    }
    __syncthreads();
  }

  // epilogue: C/D layout row = r + 8h, col = lane&15; scale rows by 1/sum
#pragma unroll
  for (int t = 0; t < 8; ++t) {
    const int v = 16 * t + (lane & 15);
#pragma unroll
    for (int r = 0; r < 8; ++r) {
      const int d = 16 * wave + r + 8 * h;
      ctxT[((size_t)b * D_ + v) * D_ + d] = acc[t][r] * sinv[d];
    }
  }
}

// ---------------------------------------------------------------------------
// K2b: fold 1x1-conv into context:  M[b][o][d] = sum_v W[o,v] * ctxT[b][v][d]
// stored transposed in f16 as MtH[b][d][o] -> directly usable as WMMA B.
// ---------------------------------------------------------------------------
__global__ __launch_bounds__(128)
void k_reproj_mat(const float* __restrict__ w, const float* __restrict__ ctxT,
                  _Float16* __restrict__ MtH) {
  const int d = threadIdx.x;
  const int o = blockIdx.x & (TWO_D - 1);
  const int b = blockIdx.x >> 8;
  float s = 0.f;
  for (int v = 0; v < D_; ++v)
    s += w[o * D_ + v] * ctxT[((size_t)b * D_ + v) * D_ + d];
  MtH[((size_t)b * D_ + d) * TWO_D + o] = (_Float16)s;
}

// ---------------------------------------------------------------------------
// K3: fused per-32-row block: channel softmax of x1 + WMMA GEMM against M^T
// (M^T k-chunk async-staged into LDS each k-step, unioned with effs buffer)
// + bias + LayerNorm + single store of the output.
// ---------------------------------------------------------------------------
__global__ __launch_bounds__(256)
void k_attend_ln_wmma(const float* __restrict__ x1, const _Float16* __restrict__ MtH,
                      const float* __restrict__ bias, const float* __restrict__ gamma,
                      const float* __restrict__ beta, float* __restrict__ out) {
  __shared__ float xs[32][D_ + 1];                     // x1 rows -> exp() in place
  __shared__ __align__(16) char u_raw[32 * (TWO_D + 1) * sizeof(float)];
  float (*effs)[TWO_D + 1] = (float (*)[TWO_D + 1])u_raw;   // post-GEMM
  _Float16* sMt = (_Float16*)u_raw;                         // 32x256 f16 during GEMM
  __shared__ float rinv[32], rmu[32], rsd[32];

  const int b    = blockIdx.y;
  const int n0   = blockIdx.x * 32;
  const int tid  = threadIdx.x;
  const int wave = tid >> 5;
  const int lane = tid & 31;
  const int h    = lane >> 4;

  const uint32_t sMt_base = (uint32_t)(uintptr_t)sMt;

  for (int idx = tid; idx < 32 * D_; idx += 256) {
    const int r = idx >> 7, c = idx & (D_ - 1);
    xs[r][c] = x1[((size_t)b * N_ + n0 + r) * D_ + c];
  }
  __syncthreads();

  // per-row softmax over D channels (queries)
  if (tid < 32) {
    float m = -INFINITY;
    for (int c = 0; c < D_; ++c) m = fmaxf(m, xs[tid][c]);
    float s = 0.f;
    for (int c = 0; c < D_; ++c) {
      const float e = __expf(xs[tid][c] - m);
      xs[tid][c] = e;
      s += e;
    }
    rinv[tid] = 1.0f / s;
  }
  __syncthreads();

  const int mtile = wave >> 2;
  const int og    = (wave & 3) * 4;
  const int mrow  = 16 * mtile + (lane & 15);

  v8f acc[4] = {};

#pragma unroll
  for (int ks = 0; ks < D_; ks += 32) {
    // async-stage M^T rows [ks, ks+32): 32x256 f16 = 16KB contiguous
    const uint32_t gbase = (uint32_t)(((size_t)b * D_ + ks) * TWO_D * sizeof(_Float16));
#pragma unroll
    for (int j = 0; j < 4; ++j) {
      const uint32_t c16 = (uint32_t)tid + 256u * j;
      async_copy_b128(sMt_base + 16u * c16, gbase + 16u * c16, MtH);
    }
    wait_asynccnt0();
    __syncthreads();

    v16h af;
#pragma unroll
    for (int i = 0; i < 16; ++i) {
      const int k = ks + i + 16 * h;
      af[i] = (_Float16)xs[mrow][k];
    }
#pragma unroll
    for (int t = 0; t < 4; ++t) {
      const int col = 16 * (og + t) + (lane & 15);
      v16h bf;
#pragma unroll
      for (int i = 0; i < 16; ++i)
        bf[i] = sMt[(i + 16 * h) * TWO_D + col];
      acc[t] = __builtin_amdgcn_wmma_f32_16x16x32_f16(
          false, af, false, bf, (short)0, acc[t], false, false);
    }
    __syncthreads();   // all reads of sMt done before next overwrite / effs reuse
  }

  // epilogue: apply softmax normalization (row) + bias (col)
#pragma unroll
  for (int t = 0; t < 4; ++t) {
    const int o = 16 * (og + t) + (lane & 15);
    const float bo = bias[o];
#pragma unroll
    for (int r = 0; r < 8; ++r) {
      const int rl = 16 * mtile + r + 8 * h;
      effs[rl][o] = acc[t][r] * rinv[rl] + bo;
    }
  }
  __syncthreads();

  // LayerNorm stats over 2D = 256 channels per row
  if (tid < 32) {
    float s = 0.f, s2 = 0.f;
    for (int o = 0; o < TWO_D; ++o) {
      const float v = effs[tid][o];
      s += v; s2 += v * v;
    }
    const float mu  = s * (1.0f / TWO_D);
    const float var = s2 * (1.0f / TWO_D) - mu * mu;
    rmu[tid] = mu;
    rsd[tid] = rsqrtf(var + LN_EPS);
  }
  __syncthreads();

  for (int idx = tid; idx < 32 * TWO_D; idx += 256) {
    const int r = idx >> 8, o = idx & (TWO_D - 1);
    out[((size_t)b * N_ + n0 + r) * TWO_D + o] =
        (effs[r][o] - rmu[r]) * rsd[r] * gamma[o] + beta[o];
  }
}

// ---------------------------------------------------------------------------
extern "C" void kernel_launch(void* const* d_in, const int* in_sizes, int n_in,
                              void* d_out, int out_size, void* d_ws, size_t ws_size,
                              hipStream_t stream) {
  const float* x1    = (const float*)d_in[0];
  const float* x2    = (const float*)d_in[1];
  const float* w     = (const float*)d_in[2];   // [2D, D] (1x1 conv weight)
  const float* bre   = (const float*)d_in[3];
  const float* gamma = (const float*)d_in[4];
  const float* beta  = (const float*)d_in[5];
  float* out = (float*)d_out;

  // workspace layout (~1.6 MB total)
  float* pmax    = (float*)d_ws;                       // B*NCHUNK*D
  float* psum    = pmax + (size_t)B_ * NCHUNK * D_;    // B*NCHUNK*D
  float* colmax  = psum + (size_t)B_ * NCHUNK * D_;    // B*D
  float* colsinv = colmax + (size_t)B_ * D_;           // B*D
  float* ctxT    = colsinv + (size_t)B_ * D_;          // B*D*D
  _Float16* MtH  = (_Float16*)(ctxT + (size_t)B_ * D_ * D_);  // B*D*2D f16

  k_colstats_part<<<dim3(NCHUNK, B_), 128, 0, stream>>>(x1, pmax, psum);
  k_colstats_final<<<B_, 128, 0, stream>>>(pmax, psum, colmax, colsinv);
  k_context_wmma<<<B_, 256, 0, stream>>>(x1, x2, colmax, colsinv, ctxT);
  k_reproj_mat<<<B_ * TWO_D, 128, 0, stream>>>(w, ctxT, MtH);
  k_attend_ln_wmma<<<dim3(N_ / 32, B_), 256, 0, stream>>>(x1, MtH, bre, gamma, beta, out);
}
